// GPTModel_2207613190569
// MI455X (gfx1250) — compile-verified, matrix operand order checked by reading, not run on hardware
//
#include <hip/hip_runtime.h>
#include <hip/hip_bf16.h>
#include <math.h>

#define NLAYER 4
#define NHEAD  12
#define DM     768
#define SEQ    2048
#define VOCAB  50257
#define BT     2
#define DH     64
#define DFF    3072
#define MTOK   (BT * SEQ)      // 4096
#define QKVW   (3 * DM)        // 2304
#define SSTRIDE 2052           // 2048 + 4 pad (bank spread)

typedef __attribute__((ext_vector_type(16))) _Float16 v16h;
typedef __attribute__((ext_vector_type(4)))  _Float16 v4h;
typedef __attribute__((ext_vector_type(8)))  float    v8f;

// ---------------- WMMA fragment gathers (ISA 7.12.2 layouts) ----------------
// A or B frag: 16-wide (M or N) index is lane&15, K contiguous in memory with
// row stride `stride`. elems 0..7 = K[koff..koff+7], elems 8..15 =
// K[koff+16..koff+23], koff = (lane>=16 ? 8 : 0) + k0.
__device__ __forceinline__ v16h frag_rowK_f16(const _Float16* base, int stride, int lane) {
  int row = lane & 15, koff = (lane >> 4) << 3;
  const _Float16* p = base + row * stride + koff;
  v16h r;
#pragma unroll
  for (int i = 0; i < 8; ++i) { r[i] = p[i]; r[i + 8] = p[i + 16]; }
  return r;
}

// f32 source, rows 16B-aligned (stride % 4 == 0, base 16B aligned): float4 loads.
__device__ __forceinline__ v16h frag_rowK_f32(const float* base, int stride, int lane, int k0) {
  int row = lane & 15, koff = ((lane >> 4) << 3) + k0;
  const float* p = base + (size_t)row * stride + koff;
  float4 a0 = *(const float4*)(p);
  float4 a1 = *(const float4*)(p + 4);
  float4 b0 = *(const float4*)(p + 16);
  float4 b1 = *(const float4*)(p + 20);
  v16h r;
  r[0]  = (_Float16)a0.x; r[1]  = (_Float16)a0.y; r[2]  = (_Float16)a0.z; r[3]  = (_Float16)a0.w;
  r[4]  = (_Float16)a1.x; r[5]  = (_Float16)a1.y; r[6]  = (_Float16)a1.z; r[7]  = (_Float16)a1.w;
  r[8]  = (_Float16)b0.x; r[9]  = (_Float16)b0.y; r[10] = (_Float16)b0.z; r[11] = (_Float16)b0.w;
  r[12] = (_Float16)b1.x; r[13] = (_Float16)b1.y; r[14] = (_Float16)b1.z; r[15] = (_Float16)b1.w;
  return r;
}

// B frag where N (lane&15) is contiguous and K strides by kstride. Batch the
// 16 independent strided loads before converting (single wait, not 16).
__device__ __forceinline__ v16h fragB_T_f32(const float* base, int kstride, int lane, int k0) {
  int n = lane & 15, koff = ((lane >> 4) << 3) + k0;
  const float* p = base + n + (size_t)koff * kstride;
  float tmp[16];
#pragma unroll
  for (int i = 0; i < 8; ++i) {
    tmp[i]     = p[(size_t)i * kstride];
    tmp[i + 8] = p[(size_t)(i + 16) * kstride];
  }
  v16h r;
#pragma unroll
  for (int i = 0; i < 16; ++i) r[i] = (_Float16)tmp[i];
  return r;
}

__device__ __forceinline__ v8f wmma_f16(v16h a, v16h b, v8f c) {
  return __builtin_amdgcn_wmma_f32_16x16x32_f16(false, a, false, b, (short)0, c, false, false);
}

__device__ __forceinline__ float gelu_exact(float v) {
  return 0.5f * v * (1.0f + erff(v * 0.70710678118654752f));
}

// ---------------- Embedding ----------------
__global__ void embed_kernel(const int* __restrict__ feats, const float* __restrict__ tok,
                             const float* __restrict__ pos, float* __restrict__ x) {
  size_t i = (size_t)blockIdx.x * 256 + threadIdx.x;
  if (i >= (size_t)MTOK * DM) return;
  int e = (int)(i % DM);
  int t = (int)(i / DM);
  int c = t % SEQ;
  x[i] = tok[(size_t)feats[t] * DM + e] + pos[(size_t)c * DM + e];
}

// ---------------- LayerNorm (1 block per row of 768) ----------------
__global__ void ln_kernel(const float* __restrict__ x, const float* __restrict__ w,
                          const float* __restrict__ b, float* __restrict__ out) {
  int row = blockIdx.x, t = threadIdx.x;
  __shared__ float s1[256], s2[256];
  const float* xr = x + (size_t)row * DM;
  float a0 = xr[t], a1 = xr[t + 256], a2 = xr[t + 512];
  s1[t] = a0 + a1 + a2;
  s2[t] = a0 * a0 + a1 * a1 + a2 * a2;
  __syncthreads();
#pragma unroll
  for (int o = 128; o > 0; o >>= 1) {
    if (t < o) { s1[t] += s1[t + o]; s2[t] += s2[t + o]; }
    __syncthreads();
  }
  float mu = s1[0] * (1.0f / DM);
  float var = s2[0] * (1.0f / DM) - mu * mu;
  float inv = rsqrtf(var + 1e-5f);
  float* orow = out + (size_t)row * DM;
  orow[t]       = (a0 - mu) * inv * w[t]       + b[t];
  orow[t + 256] = (a1 - mu) * inv * w[t + 256] + b[t + 256];
  orow[t + 512] = (a2 - mu) * inv * w[t + 512] + b[t + 512];
}

// ---------------- Generic WMMA GEMM (LDS double-buffered) ----------------
// out[M,N] = act(A[M,K] @ W[K,N] + bias) (+ residual).
// M multiple of 128, K multiple of 32 and 4 (A rows 16B aligned).
// Block: 256 threads = 8 waves (4 in M x 2 in N); tile 128x128; K step 32.
// Each wave: 32x64 output = 8 accumulators, 8 WMMAs per K step (B reused).
__global__ __launch_bounds__(256) void gemm_wmma(
    const float* __restrict__ A, const float* __restrict__ W,
    const float* __restrict__ bias, const float* __restrict__ residual,
    float* __restrict__ out, int M, int N, int K, int act) {
  __shared__ _Float16 As[2][128][40];
  __shared__ _Float16 Bs[2][128][40];
  int t = threadIdx.x, lane = t & 31, wid = t >> 5;
  int wm = wid & 3, wn = wid >> 2;
  int gm0 = blockIdx.y * 128;
  int bn0 = blockIdx.x * 128;

  // Per-thread staging coordinates are loop-invariant:
  //   A: rows rA+32i (i<4), cols cA..cA+3
  //   B: n = nB, k-run kkB..kkB+15 (contiguous -> packed ds_store_b64 drain)
  int rA = t >> 3;                 // 0..31
  int cA = (t & 7) * 4;            // 0..28
  const float* pA = A + (size_t)(gm0 + rA) * K + cA;
  int nB = t & 127;                // 0..127
  int kkB = (t >> 7) * 16;         // 0 or 16
  int gn = bn0 + nB;
  bool nvalid = gn < N;
  int gnc = nvalid ? gn : (N - 1); // clamp: loads always in-bounds
  const float* pW = W + (size_t)kkB * N + gnc;

  float4 ta[4];
  float tb[16];
  // ---- prologue: stage tile k0=0 into buffer 0 ----
#pragma unroll
  for (int i = 0; i < 4; ++i) ta[i] = *(const float4*)(pA + (size_t)(i * 32) * K);
#pragma unroll
  for (int i = 0; i < 16; ++i) tb[i] = pW[(size_t)i * N];
#pragma unroll
  for (int i = 0; i < 4; ++i) {
    v4h h4;
    h4[0] = (_Float16)ta[i].x; h4[1] = (_Float16)ta[i].y;
    h4[2] = (_Float16)ta[i].z; h4[3] = (_Float16)ta[i].w;
    *(v4h*)&As[0][rA + i * 32][cA] = h4;
  }
#pragma unroll
  for (int c = 0; c < 4; ++c) {
    v4h h4;
#pragma unroll
    for (int u = 0; u < 4; ++u)
      h4[u] = nvalid ? (_Float16)tb[4 * c + u] : (_Float16)0.0f;
    *(v4h*)&Bs[0][nB][kkB + 4 * c] = h4;
  }
  __syncthreads();

  v8f acc[2][4];
#pragma unroll
  for (int g = 0; g < 2; ++g)
#pragma unroll
    for (int j = 0; j < 4; ++j) acc[g][j] = (v8f){0, 0, 0, 0, 0, 0, 0, 0};

  int buf = 0;
  for (int k0 = 0; k0 < K; k0 += 32) {
    bool more = (k0 + 32) < K;          // block-uniform
    if (more) {                         // prefetch next tile into registers
      int kn = k0 + 32;
#pragma unroll
      for (int i = 0; i < 4; ++i) ta[i] = *(const float4*)(pA + (size_t)(i * 32) * K + kn);
#pragma unroll
      for (int i = 0; i < 16; ++i) tb[i] = pW[(size_t)(kn + i) * N];
    }
    // compute on current buffer (overlaps with in-flight global loads)
    v16h a0 = frag_rowK_f16(&As[buf][wm * 32][0], 40, lane);
    v16h a1 = frag_rowK_f16(&As[buf][wm * 32 + 16][0], 40, lane);
#pragma unroll
    for (int j = 0; j < 4; ++j) {
      v16h bfr = frag_rowK_f16(&Bs[buf][wn * 64 + j * 16][0], 40, lane);
      acc[0][j] = wmma_f16(a0, bfr, acc[0][j]);
      acc[1][j] = wmma_f16(a1, bfr, acc[1][j]);
    }
    if (more) {                         // drain prefetch into other buffer
#pragma unroll
      for (int i = 0; i < 4; ++i) {
        v4h h4;
        h4[0] = (_Float16)ta[i].x; h4[1] = (_Float16)ta[i].y;
        h4[2] = (_Float16)ta[i].z; h4[3] = (_Float16)ta[i].w;
        *(v4h*)&As[buf ^ 1][rA + i * 32][cA] = h4;
      }
#pragma unroll
      for (int c = 0; c < 4; ++c) {
        v4h h4;
#pragma unroll
        for (int u = 0; u < 4; ++u)
          h4[u] = nvalid ? (_Float16)tb[4 * c + u] : (_Float16)0.0f;
        *(v4h*)&Bs[buf ^ 1][nB][kkB + 4 * c] = h4;
      }
    }
    __syncthreads();
    buf ^= 1;
  }

  int nlane = lane & 15;
  int mb = (lane >> 4) * 8;
#pragma unroll
  for (int g = 0; g < 2; ++g) {
#pragma unroll
    for (int j = 0; j < 4; ++j) {
      int n = bn0 + wn * 64 + j * 16 + nlane;
      if (n < N) {
        float bv = bias ? bias[n] : 0.0f;
#pragma unroll
        for (int r = 0; r < 8; ++r) {
          int m = gm0 + wm * 32 + g * 16 + mb + r;
          float v = acc[g][j][r] + bv;
          if (act == 1) v = gelu_exact(v);
          if (residual) v += residual[(size_t)m * N + n];
          out[(size_t)m * N + n] = v;
        }
      }
    }
  }
}

// ---------------- Fused causal attention ----------------
// Grid (SEQ/16, NHEAD, BT), block 128 (4 waves). Scores for a 16-query tile
// live in dynamic LDS (16 x SSTRIDE f32 = 128 KB, fits 320 KB WGP LDS).
__global__ __launch_bounds__(128) void attn_kernel(const float* __restrict__ qkv,
                                                   float* __restrict__ y) {
  extern __shared__ float sS[];
  int q0 = blockIdx.x * 16;
  int h = blockIdx.y, b = blockIdx.z;
  int t = threadIdx.x, lane = t & 31, wid = t >> 5;

  const float* base = qkv + (size_t)b * SEQ * QKVW;
  const float* Qb = base + h * DH;
  const float* Kb = base + DM + h * DH;
  const float* Vb = base + 2 * DM + h * DH;

  // Q frags for rows q0..q0+15 (dh=64 -> two K-chunks)
  v16h qa0 = frag_rowK_f32(Qb + (size_t)q0 * QKVW, QKVW, lane, 0);
  v16h qa1 = frag_rowK_f32(Qb + (size_t)q0 * QKVW, QKVW, lane, 32);

  int jmax = q0 >> 4;                 // key tiles 0..jmax
  int nlane = lane & 15;
  int mb = (lane >> 4) * 8;
  for (int j = wid; j <= jmax; j += 4) {   // wave-uniform -> EXEC all ones
    v16h kb0 = frag_rowK_f32(Kb + (size_t)(j * 16) * QKVW, QKVW, lane, 0);
    v16h kb1 = frag_rowK_f32(Kb + (size_t)(j * 16) * QKVW, QKVW, lane, 32);
    v8f c = (v8f){0, 0, 0, 0, 0, 0, 0, 0};
    c = wmma_f16(qa0, kb0, c);
    c = wmma_f16(qa1, kb1, c);
#pragma unroll
    for (int r = 0; r < 8; ++r)
      sS[(size_t)(mb + r) * SSTRIDE + j * 16 + nlane] = c[r] * 0.125f; // 1/sqrt(64)
  }
  __syncthreads();

  // Causal softmax: 8 threads per row (rows 0..15), shuffle-reduce width 8.
  int row = t >> 3, sub = t & 7;
  int query = q0 + row;
  int kvLen = query + 1;
  int kvPad = ((q0 + 16 + 31) >> 5) << 5;
  float* srow = sS + (size_t)row * SSTRIDE;
  float mx = -3.4e38f;
  for (int k = sub; k < kvLen; k += 8) mx = fmaxf(mx, srow[k]);
#pragma unroll
  for (int o = 4; o > 0; o >>= 1) mx = fmaxf(mx, __shfl_xor(mx, o, 8));
  float sum = 0.0f;
  for (int k = sub; k < kvLen; k += 8) { float p = __expf(srow[k] - mx); srow[k] = p; sum += p; }
#pragma unroll
  for (int o = 4; o > 0; o >>= 1) sum += __shfl_xor(sum, o, 8);
  float inv = 1.0f / sum;
  for (int k = sub; k < kvLen; k += 8) srow[k] *= inv;
  for (int k = sub; k < kvPad; k += 8) if (k >= kvLen) srow[k] = 0.0f;
  __syncthreads();

  // P @ V: each wave owns 16 of the 64 head-dim cols; K chunks of 32.
  v8f o = (v8f){0, 0, 0, 0, 0, 0, 0, 0};
  const float* Vcol = Vb + wid * 16;
  for (int kc = 0; kc < kvPad; kc += 32) {
    v16h pa = frag_rowK_f32(sS, SSTRIDE, lane, kc);
    v16h vb = fragB_T_f32(Vcol, QKVW, lane, kc);
    o = wmma_f16(pa, vb, o);
  }
  int n = h * DH + wid * 16 + nlane;
#pragma unroll
  for (int r = 0; r < 8; ++r)
    y[(size_t)(b * SEQ + q0 + mb + r) * DM + n] = o[r];
}

// ---------------- Loss ----------------
__global__ void zero_accum(float* a) { if (threadIdx.x == 0 && blockIdx.x == 0) a[0] = 0.0f; }

__global__ void loss_rows(const float* __restrict__ logits, const int* __restrict__ targets,
                          float* __restrict__ accum) {
  int row = blockIdx.x, t = threadIdx.x;
  const float* lr = logits + (size_t)row * VOCAB;
  __shared__ float red[256];
  float mx = -3.4e38f;
  for (int k = t; k < VOCAB; k += 256) mx = fmaxf(mx, lr[k]);
  red[t] = mx; __syncthreads();
#pragma unroll
  for (int o = 128; o > 0; o >>= 1) { if (t < o) red[t] = fmaxf(red[t], red[t + o]); __syncthreads(); }
  mx = red[0]; __syncthreads();
  float sum = 0.0f;
  for (int k = t; k < VOCAB; k += 256) sum += __expf(lr[k] - mx);
  red[t] = sum; __syncthreads();
#pragma unroll
  for (int o = 128; o > 0; o >>= 1) { if (t < o) red[t] += red[t + o]; __syncthreads(); }
  if (t == 0) {
    int tgt = targets[row];
    if (tgt >= 0) {
      float lse = mx + __logf(red[0]);
      atomicAdd(accum, lse - lr[tgt]);
    }
  }
}

__global__ void finalize_loss(const float* __restrict__ accum, float* __restrict__ loss_out) {
  if (threadIdx.x == 0 && blockIdx.x == 0) loss_out[0] = accum[0] / (float)MTOK;
}

// ---------------- Host driver ----------------
extern "C" void kernel_launch(void* const* d_in, const int* in_sizes, int n_in,
                              void* d_out, int out_size, void* d_ws, size_t ws_size,
                              hipStream_t stream) {
  (void)in_sizes; (void)n_in; (void)out_size; (void)ws_size;
  const int*   feats    = (const int*)d_in[0];
  const int*   targets  = (const int*)d_in[1];
  const float* tok      = (const float*)d_in[2];
  const float* pos      = (const float*)d_in[3];
  const float* ln1_w    = (const float*)d_in[4];
  const float* ln1_b    = (const float*)d_in[5];
  const float* attn_w   = (const float*)d_in[6];
  const float* attn_b   = (const float*)d_in[7];
  const float* proj_w   = (const float*)d_in[8];
  const float* proj_b   = (const float*)d_in[9];
  const float* ln2_w    = (const float*)d_in[10];
  const float* ln2_b    = (const float*)d_in[11];
  const float* ffn_w1   = (const float*)d_in[12];
  const float* ffn_b1   = (const float*)d_in[13];
  const float* ffn_w2   = (const float*)d_in[14];
  const float* ffn_b2   = (const float*)d_in[15];
  const float* lnf_w    = (const float*)d_in[16];
  const float* lnf_b    = (const float*)d_in[17];
  const float* final_w  = (const float*)d_in[18];

  float* logits = (float*)d_out;
  float* ws = (float*)d_ws;
  const size_t XE = (size_t)MTOK * DM;
  float* x    = ws;                        // [4096, 768]
  float* hbuf = x + XE;                    // [4096, 768]
  float* qkv  = hbuf + XE;                 // [4096, 2304]
  float* ybuf = qkv + (size_t)MTOK * QKVW; // [4096, 768]
  float* ffn  = ybuf + XE;                 // [4096, 3072]
  float* accum = ffn + (size_t)MTOK * DFF; // [1]

  // embed
  {
    size_t total = (size_t)MTOK * DM;
    embed_kernel<<<dim3((unsigned)((total + 255) / 256)), dim3(256), 0, stream>>>(feats, tok, pos, x);
  }

  const size_t attn_shmem = (size_t)16 * SSTRIDE * sizeof(float); // 128 KB (<320 KB WGP LDS)

  for (int l = 0; l < NLAYER; ++l) {
    const float* aW  = attn_w + (size_t)l * DM * QKVW;
    const float* aB  = attn_b + (size_t)l * QKVW;
    const float* pW  = proj_w + (size_t)l * DM * DM;
    const float* pB  = proj_b + (size_t)l * DM;
    const float* f1W = ffn_w1 + (size_t)l * DM * DFF;
    const float* f1B = ffn_b1 + (size_t)l * DFF;
    const float* f2W = ffn_w2 + (size_t)l * DFF * DM;
    const float* f2B = ffn_b2 + (size_t)l * DM;

    // h = LN1(x)
    ln_kernel<<<dim3(MTOK), dim3(256), 0, stream>>>(x, ln1_w + (size_t)l * DM, ln1_b + (size_t)l * DM, hbuf);
    // qkv = h @ attn_w + attn_b
    gemm_wmma<<<dim3((QKVW + 127) / 128, MTOK / 128), dim3(256), 0, stream>>>(
        hbuf, aW, aB, nullptr, qkv, MTOK, QKVW, DM, 0);
    // y = attention(qkv)
    attn_kernel<<<dim3(SEQ / 16, NHEAD, BT), dim3(128), attn_shmem, stream>>>(qkv, ybuf);
    // x = y @ proj_w + proj_b + x
    gemm_wmma<<<dim3((DM + 127) / 128, MTOK / 128), dim3(256), 0, stream>>>(
        ybuf, pW, pB, x, x, MTOK, DM, DM, 0);
    // h = LN2(x)
    ln_kernel<<<dim3(MTOK), dim3(256), 0, stream>>>(x, ln2_w + (size_t)l * DM, ln2_b + (size_t)l * DM, hbuf);
    // ffn = gelu(h @ w1 + b1)
    gemm_wmma<<<dim3((DFF + 127) / 128, MTOK / 128), dim3(256), 0, stream>>>(
        hbuf, f1W, f1B, nullptr, ffn, MTOK, DFF, DM, 1);
    // x = ffn @ w2 + b2 + x
    gemm_wmma<<<dim3((DM + 127) / 128, MTOK / 128), dim3(256), 0, stream>>>(
        ffn, f2W, f2B, x, x, MTOK, DM, DFF, 0);
  }

  // h = LN_f(x); logits = h @ final_w
  ln_kernel<<<dim3(MTOK), dim3(256), 0, stream>>>(x, lnf_w, lnf_b, hbuf);
  gemm_wmma<<<dim3((VOCAB + 127) / 128, MTOK / 128), dim3(256), 0, stream>>>(
      hbuf, final_w, nullptr, nullptr, logits, MTOK, VOCAB, DM, 0);

  // loss
  zero_accum<<<dim3(1), dim3(32), 0, stream>>>(accum);
  loss_rows<<<dim3(MTOK), dim3(256), 0, stream>>>(logits, targets, accum);
  finalize_loss<<<dim3(1), dim3(32), 0, stream>>>(accum, logits + (size_t)MTOK * VOCAB);
}
